// XASStructureV2_41841571397766
// MI455X (gfx1250) — matrix operand order repaired and checked by reading, not compiled, not run on hardware
//
#include <hip/hip_runtime.h>
#include <hip/hip_bf16.h>
#include <cstdint>

// ---------------- problem dimensions (fixed by the reference) ----------------
#define NN     8192      // nodes
#define EE     131072    // edges
#define BBATCH 16        // graphs
#define LSPEC  100       // spectrum length
#define HH     256       // hidden
#define ELEN   32        // energy levels
#define EDIM   864       // edge feature dim
#define NLAY   3
#define DIN    (2*HH+EDIM)
#define BLROWS (BBATCH*LSPEC)   // 1600

// ---------------- CDNA5 WMMA / TDM types ----------------
typedef __bf16 bf16;
typedef __attribute__((ext_vector_type(16))) bf16  v16bf;
typedef __attribute__((ext_vector_type(8)))  float v8f;
typedef __attribute__((ext_vector_type(4)))  unsigned int v4u;
typedef __attribute__((ext_vector_type(8)))  int v8i;
typedef __attribute__((ext_vector_type(4)))  int v4i;

union FragAB { uint4 q[2]; v16bf v; };
union FragU  { unsigned short s[16]; v16bf v; };

__device__ __forceinline__ unsigned short f2bf(float f) {
    unsigned int u = __builtin_bit_cast(unsigned int, f);
    unsigned int r = u + 0x7FFFu + ((u >> 16) & 1u);   // round-to-nearest-even
    return (unsigned short)(r >> 16);
}

__device__ __forceinline__ v8f wmma_bf16(v16bf a, v16bf b, v8f c) {
    return __builtin_amdgcn_wmma_f32_16x16x32_bf16(
        false, a, false, b, (short)0, c, false, false);
}

// =====================================================================
// Generic bf16 GEMM:  C[M,Nc] = scale * (A[M,K] @ B[K,Nc]) + bias
// A row-major bf16 (lda=K); Bt column-major: Bt[n*K + k].
// Block = 128 threads = 4 waves; tile 16 rows x 128 cols (2 N-tiles/wave).
// Double-buffered LDS A-tile: one barrier per 32-wide K step.
// Optional ragged-graph mask (scores): C[m,n] = -1e30 if node_graph[n]!=m/L.
// =====================================================================
__global__ void __launch_bounds__(128)
k_gemm_bf16(const unsigned short* __restrict__ A,
            const unsigned short* __restrict__ Bt,
            const float* __restrict__ bias,
            float* __restrict__ C,
            int M, int Nc, int K, float scale,
            const int* __restrict__ node_graph, int Lmask)
{
    __shared__ __align__(16) unsigned short aSh[2][16 * 32];

    const int wave = threadIdx.x >> 5;
    const int lane = threadIdx.x & 31;
    const int hi   = lane >> 4;
    const int lm   = lane & 15;
    const int m0   = blockIdx.x * 16;
    const int nb   = blockIdx.y * 128 + wave * 32;    // wave's 32-col strip
    const int nA   = nb + lm;
    const int nB   = nb + 16 + lm;

    v8f acc0, acc1;
    for (int r = 0; r < 8; ++r) { acc0[r] = 0.0f; acc1[r] = 0.0f; }

    const int arow = threadIdx.x >> 3;          // 0..15
    const int acol = (threadIdx.x & 7) * 4;     // 0..28
    const unsigned short* aRow = A + (size_t)(m0 + arow) * K + acol;

    // preload K-step 0 into buffer 0
    uint2 areg = *(const uint2*)aRow;
    *(uint2*)&aSh[0][arow * 32 + acol] = areg;
    int cur = 0;

    for (int k0 = 0; k0 < K; k0 += 32) {
        __syncthreads();                         // aSh[cur] ready
        const bool has_next = (k0 + 32) < K;
        if (has_next) areg = *(const uint2*)(aRow + k0 + 32);

        FragAB fa, fb0, fb1;
        const int ab = lm * 32 + hi * 8;         // ISA 16-bit A layout
        fa.q[0] = *(const uint4*)&aSh[cur][ab];
        fa.q[1] = *(const uint4*)&aSh[cur][ab + 16];
        const unsigned short* b0 = Bt + (size_t)nA * K + k0 + hi * 8;
        const unsigned short* b1 = Bt + (size_t)nB * K + k0 + hi * 8;
        fb0.q[0] = *(const uint4*)b0;  fb0.q[1] = *(const uint4*)(b0 + 16);
        fb1.q[0] = *(const uint4*)b1;  fb1.q[1] = *(const uint4*)(b1 + 16);

        acc0 = wmma_bf16(fa.v, fb0.v, acc0);
        acc1 = wmma_bf16(fa.v, fb1.v, acc1);

        if (has_next) *(uint2*)&aSh[cur ^ 1][arow * 32 + acol] = areg;
        cur ^= 1;
    }

    for (int t = 0; t < 2; ++t) {
        v8f& acc  = t ? acc1 : acc0;
        const int n = t ? nB : nA;
        const float bv = bias ? bias[n] : 0.0f;
        for (int r = 0; r < 8; ++r) {
            const int m = m0 + r + hi * 8;       // C/D VGPR layout
            float v = acc[r] * scale + bv;
            if (node_graph) {
                if (node_graph[n] != (m / Lmask)) v = -1e30f;
            }
            C[(size_t)m * Nc + n] = v;
        }
    }
}

// =====================================================================
// Edge message kernel. One block = 16 edges x full H=256.
// The 16x864 f32 sbhf tile is DMA'd into LDS by the Tensor Data Mover,
// then the K-loop runs barrier-free: LDS f32 -> bf16 frags -> WMMA.
// msg = sbhf@Wmid + Asrc[src] + Adst[dst] (b_agg folded into Asrc),
// scaled by r^exp_p, atomically scattered to agg[dst].
// =====================================================================
__global__ void __launch_bounds__(256)
k_edge_msg(const float* __restrict__ sbhf,            // [E,864] f32
           const unsigned short* __restrict__ Wmid_t, // [256][864] bf16 col-major
           const float* __restrict__ Asrc,            // [N,256]
           const float* __restrict__ Adst,            // [N,256]
           const int* __restrict__ esrc,
           const int* __restrict__ edst,
           const float* __restrict__ eradius,
           const float* __restrict__ exp_p, int layer,
           float* __restrict__ agg)                   // [N,256]
{
    __shared__ __align__(16) float sF[16 * EDIM];     // 55296 B TDM landing pad
    __shared__ int   sSrc[16], sDst[16];
    __shared__ float sW[16];

    const int e0   = blockIdx.x * 16;
    const int wave = threadIdx.x >> 5;
    const int lane = threadIdx.x & 31;
    const int hi   = lane >> 4;
    const int lm   = lane & 15;

    if (threadIdx.x < 16) {
        const int e = e0 + threadIdx.x;
        sSrc[threadIdx.x] = esrc[e];
        sDst[threadIdx.x] = edst[e];
        sW[threadIdx.x]   = __expf(exp_p[layer] * __logf(eradius[e]));  // r^p
    }

    // ---- Tensor Data Mover: DMA 16 rows x 864 f32 into LDS (wave 0 only) ----
    if (wave == 0) {
        const unsigned long long ga =
            (unsigned long long)(const void*)(sbhf + (size_t)e0 * EDIM);
        const unsigned int lds = (unsigned int)(size_t)(void*)sF; // LDS aperture offset
        v4u g0;                                   // D# group 0 (ISA 8.3)
        g0[0] = 0x1u;                             // count=1, user mode
        g0[1] = lds;                              // lds_addr
        g0[2] = (unsigned int)(ga & 0xFFFFFFFFull);
        g0[3] = (unsigned int)((ga >> 32) & 0x01FFFFFFull) | 0x80000000u; // type=2
        v8i g1;                                   // D# group 1 (ISA 8.4)
        g1[0] = 0x00020000;                       // wg_mask=0, data_size=4B
        g1[1] = (int)(((unsigned)EDIM & 0xFFFFu) << 16);   // tensor_dim0 lo16
        g1[2] = (int)((EDIM >> 16) | (16u << 16));         // dim0 hi16 | tensor_dim1 lo16
        g1[3] = (int)(((unsigned)EDIM) << 16);             // dim1 hi16=0 | tile_dim0=864
        g1[4] = 16;                               // tile_dim1=16, tile_dim2=0
        g1[5] = EDIM;                             // tensor_dim0_stride lo32
        g1[6] = 0;                                // stride0 hi16, stride1 lo16
        g1[7] = 0;
        v4i g2; g2[0]=0; g2[1]=0; g2[2]=0; g2[3]=0;   // dims 2/3 unused
        v4i g3; g3[0]=0; g3[1]=0; g3[2]=0; g3[3]=0;
        v8i g4; g4[0]=0; g4[1]=0; g4[2]=0; g4[3]=0;   // extra group (clang-23 6-arg form)
        g4[4]=0; g4[5]=0; g4[6]=0; g4[7]=0;
        __builtin_amdgcn_tensor_load_to_lds(g0, g1, g2, g3, g4, 0);
        __builtin_amdgcn_s_wait_tensorcnt(0);
    }
    __syncthreads();                              // tile + edge meta visible to all

    v8f acc0, acc1;
    for (int r = 0; r < 8; ++r) { acc0[r] = 0.0f; acc1[r] = 0.0f; }

    const int n0 = wave * 32 + lm;
    const int n1 = wave * 32 + 16 + lm;
    const float* aRow = sF + lm * EDIM + hi * 8;  // this lane's A row base

    for (int k0 = 0; k0 < EDIM; k0 += 32) {       // 27 iterations, no barriers
        // A fragment: f32 from LDS -> bf16 (ISA 16-bit A layout)
        const float4 x0 = *(const float4*)(aRow + k0);
        const float4 x1 = *(const float4*)(aRow + k0 + 4);
        const float4 x2 = *(const float4*)(aRow + k0 + 16);
        const float4 x3 = *(const float4*)(aRow + k0 + 20);
        FragU fa;
        fa.s[0]=f2bf(x0.x);  fa.s[1]=f2bf(x0.y);  fa.s[2]=f2bf(x0.z);  fa.s[3]=f2bf(x0.w);
        fa.s[4]=f2bf(x1.x);  fa.s[5]=f2bf(x1.y);  fa.s[6]=f2bf(x1.z);  fa.s[7]=f2bf(x1.w);
        fa.s[8]=f2bf(x2.x);  fa.s[9]=f2bf(x2.y);  fa.s[10]=f2bf(x2.z); fa.s[11]=f2bf(x2.w);
        fa.s[12]=f2bf(x3.x); fa.s[13]=f2bf(x3.y); fa.s[14]=f2bf(x3.z); fa.s[15]=f2bf(x3.w);

        FragAB fb0, fb1;
        const unsigned short* b0 = Wmid_t + (size_t)n0 * EDIM + k0 + hi * 8;
        const unsigned short* b1 = Wmid_t + (size_t)n1 * EDIM + k0 + hi * 8;
        fb0.q[0] = *(const uint4*)b0;  fb0.q[1] = *(const uint4*)(b0 + 16);
        fb1.q[0] = *(const uint4*)b1;  fb1.q[1] = *(const uint4*)(b1 + 16);

        acc0 = wmma_bf16(fa.v, fb0.v, acc0);
        acc1 = wmma_bf16(fa.v, fb1.v, acc1);
    }

    // epilogue: gather node contributions, weight, scatter-add to agg[dst]
    for (int t = 0; t < 2; ++t) {
        v8f& acc  = t ? acc1 : acc0;
        const int n = t ? n1 : n0;
        for (int r = 0; r < 8; ++r) {
            const int m = r + hi * 8;
            const int s = sSrc[m];
            const int d = sDst[m];
            const float v = (acc[r]
                             + Asrc[(size_t)s * HH + n]
                             + Adst[(size_t)d * HH + n]) * sW[m];
            atomicAdd(&agg[(size_t)d * HH + n], v);
        }
    }
}

// ---------------- pointwise / helper kernels ----------------
__global__ void k_cvt_bf16(const float* __restrict__ s, unsigned short* __restrict__ d, int n) {
    int i = blockIdx.x * blockDim.x + threadIdx.x;
    if (i < n) d[i] = f2bf(s[i]);
}
__global__ void k_transpose_cvt(const float* __restrict__ s, unsigned short* __restrict__ d,
                                int R, int C) {  // s: [R][C] row-major -> d: [C][R]
    int i = blockIdx.x * blockDim.x + threadIdx.x;
    if (i < R * C) {
        int r = i / C, c = i % C;
        d[(size_t)c * R + r] = f2bf(s[i]);
    }
}
__global__ void k_zero_f32(float* p, int n) {
    int i = blockIdx.x * blockDim.x + threadIdx.x;
    if (i < n) p[i] = 0.0f;
}
__global__ void k_fill1(float* p, int n) {
    int i = blockIdx.x * blockDim.x + threadIdx.x;
    if (i < n) p[i] = 1.0f;
}
__global__ void k_spec_x(const float* __restrict__ xr, const float* __restrict__ Ws,
                         const float* __restrict__ bs, unsigned short* __restrict__ sxb) {
    int i = blockIdx.x * blockDim.x + threadIdx.x;   // BLROWS*HH exact
    int bl = i >> 8, h = i & 255;
    sxb[i] = f2bf(xr[bl] * Ws[h] + bs[h]);
}
__global__ void k_rst(const float* __restrict__ h0, const float* __restrict__ agg,
                      const float* __restrict__ eps_p, int layer,
                      unsigned short* __restrict__ rstb, int n) {
    int i = blockIdx.x * blockDim.x + threadIdx.x;
    if (i < n) rstb[i] = f2bf((1.0f + eps_p[layer]) * h0[i] + agg[i]);
}
__global__ void k_glu(const float* __restrict__ g1, const float* __restrict__ g2,
                      float* __restrict__ hf, unsigned short* __restrict__ hb, int n) {
    int i = blockIdx.x * blockDim.x + threadIdx.x;
    if (i < n) {
        float v = g1[i] * (1.0f / (1.0f + __expf(-g2[i])));
        hf[i] = v; hb[i] = f2bf(v);
    }
}

// masked softmax over Nc columns (scores already hold -1e30 where masked)
__global__ void __launch_bounds__(256)
k_softmax_row(const float* __restrict__ S, unsigned short* __restrict__ P, int Nc) {
    __shared__ float red[256];
    const int row = blockIdx.x;
    const float* s = S + (size_t)row * Nc;
    float mx = -3.4e38f;
    for (int j = threadIdx.x; j < Nc; j += 256) mx = fmaxf(mx, s[j]);
    red[threadIdx.x] = mx; __syncthreads();
    for (int st = 128; st > 0; st >>= 1) {
        if (threadIdx.x < st) red[threadIdx.x] = fmaxf(red[threadIdx.x], red[threadIdx.x + st]);
        __syncthreads();
    }
    mx = red[0]; __syncthreads();
    float sum = 0.0f;
    for (int j = threadIdx.x; j < Nc; j += 256) sum += __expf(s[j] - mx);
    red[threadIdx.x] = sum; __syncthreads();
    for (int st = 128; st > 0; st >>= 1) {
        if (threadIdx.x < st) red[threadIdx.x] += red[threadIdx.x + st];
        __syncthreads();
    }
    const float inv = 1.0f / red[0];
    unsigned short* p = P + (size_t)row * Nc;
    for (int j = threadIdx.x; j < Nc; j += 256) p[j] = f2bf(__expf(s[j] - mx) * inv);
}

// spec_y[bl] = spec_y[bl] * dot(y[bl,:], W_enc) + b_enc
__global__ void __launch_bounds__(256)
k_spec_update(const float* __restrict__ Y, const float* __restrict__ Wenc,
              const float* __restrict__ benc, float* __restrict__ sOut) {
    __shared__ float red[256];
    const int row = blockIdx.x;
    red[threadIdx.x] = Y[(size_t)row * HH + threadIdx.x] * Wenc[threadIdx.x];
    __syncthreads();
    for (int st = 128; st > 0; st >>= 1) {
        if (threadIdx.x < st) red[threadIdx.x] += red[threadIdx.x + st];
        __syncthreads();
    }
    if (threadIdx.x == 0) sOut[row] = sOut[row] * red[0] + benc[0];
}

// =====================================================================
extern "C" void kernel_launch(void* const* d_in, const int* in_sizes, int n_in,
                              void* d_out, int out_size, void* d_ws, size_t ws_size,
                              hipStream_t stream) {
    (void)in_sizes; (void)n_in; (void)out_size; (void)ws_size;
    const float* node_energy = (const float*)d_in[0];
    const float* spec_x_raw  = (const float*)d_in[1];
    const float* edge_radius = (const float*)d_in[2];
    const float* edge_sbhf   = (const float*)d_in[3];
    const int*   edge_src    = (const int*)d_in[4];
    const int*   edge_dst    = (const int*)d_in[5];
    const int*   node_graph  = (const int*)d_in[6];
    const float* W_energy    = (const float*)d_in[7];
    const float* b_energy    = (const float*)d_in[8];
    const float* W_spec      = (const float*)d_in[9];
    const float* b_spec      = (const float*)d_in[10];
    const float* W_enc       = (const float*)d_in[11];
    const float* b_enc       = (const float*)d_in[12];
    const float* W_agg       = (const float*)d_in[13];
    const float* b_agg       = (const float*)d_in[14];
    const float* W_glu1      = (const float*)d_in[15];
    const float* b_glu1      = (const float*)d_in[16];
    const float* W_glu2      = (const float*)d_in[17];
    const float* b_glu2      = (const float*)d_in[18];
    const float* exp_p       = (const float*)d_in[19];
    const float* eps_p       = (const float*)d_in[20];
    float* out = (float*)d_out;   // [B,L] f32

    // -------- workspace layout --------
    char* ws = (char*)d_ws;
    size_t off = 0;
    auto alloc = [&](size_t bytes) -> char* {
        off = (off + 255) & ~(size_t)255;
        char* p = ws + off; off += bytes; return p;
    };
    unsigned short* neb    = (unsigned short*)alloc((size_t)NN * ELEN * 2);
    unsigned short* WEt    = (unsigned short*)alloc((size_t)HH * ELEN * 2);
    float*          h0     = (float*)         alloc((size_t)NN * HH * 4);
    unsigned short* h0b    = (unsigned short*)alloc((size_t)NN * HH * 2);
    unsigned short* sxb    = (unsigned short*)alloc((size_t)BLROWS * HH * 2);
    unsigned short* Wsrc_t = (unsigned short*)alloc((size_t)HH * HH * 2);
    unsigned short* Wdst_t = (unsigned short*)alloc((size_t)HH * HH * 2);
    unsigned short* Wmid_t = (unsigned short*)alloc((size_t)HH * EDIM * 2);
    unsigned short* Wg1t   = (unsigned short*)alloc((size_t)HH * HH * 2);
    unsigned short* Wg2t   = (unsigned short*)alloc((size_t)HH * HH * 2);
    float*          Asrc   = (float*)         alloc((size_t)NN * HH * 4);
    float*          Adst   = (float*)         alloc((size_t)NN * HH * 4);
    float*          agg    = (float*)         alloc((size_t)NN * HH * 4);
    unsigned short* rstb   = (unsigned short*)alloc((size_t)NN * HH * 2);
    float*          G1     = (float*)         alloc((size_t)NN * HH * 4);
    float*          G2     = (float*)         alloc((size_t)NN * HH * 4);
    float*          hf     = (float*)         alloc((size_t)NN * HH * 4);
    unsigned short* hb     = (unsigned short*)alloc((size_t)NN * HH * 2);
    unsigned short* hTb    = (unsigned short*)alloc((size_t)NN * HH * 2);
    float*          scores = (float*)         alloc((size_t)BLROWS * NN * 4);
    unsigned short* Pb     = (unsigned short*)alloc((size_t)BLROWS * NN * 2);
    float*          Yf     = (float*)         alloc((size_t)BLROWS * HH * 4);

    const int T = 256;
    auto g = [](int n, int t) { return (n + t - 1) / t; };

    // -------- prologue --------
    k_cvt_bf16<<<g(NN*ELEN,T), T, 0, stream>>>(node_energy, neb, NN*ELEN);
    k_transpose_cvt<<<g(ELEN*HH,T), T, 0, stream>>>(W_energy, WEt, ELEN, HH);
    // h0 = node_energy @ W_energy + b_energy   [8192,32]x[32,256]
    k_gemm_bf16<<<dim3(NN/16, HH/128), 128, 0, stream>>>(
        neb, WEt, b_energy, h0, NN, HH, ELEN, 1.0f, nullptr, 1);
    k_cvt_bf16<<<g(NN*HH,T), T, 0, stream>>>(h0, h0b, NN*HH);
    k_spec_x<<<(BLROWS*HH)/T, T, 0, stream>>>(spec_x_raw, W_spec, b_spec, sxb);
    k_fill1<<<g(BLROWS,T), T, 0, stream>>>(out, BLROWS);   // spec_y = ones

    // -------- layers --------
    for (int i = 0; i < NLAY; ++i) {
        const float* Wl = W_agg + (size_t)i * DIN * HH;
        k_transpose_cvt<<<g(HH*HH,T), T, 0, stream>>>(Wl, Wsrc_t, HH, HH);
        k_transpose_cvt<<<g(EDIM*HH,T), T, 0, stream>>>(Wl + (size_t)HH*HH, Wmid_t, EDIM, HH);
        k_transpose_cvt<<<g(HH*HH,T), T, 0, stream>>>(Wl + (size_t)(HH+EDIM)*HH, Wdst_t, HH, HH);
        k_transpose_cvt<<<g(HH*HH,T), T, 0, stream>>>(W_glu1 + (size_t)i*HH*HH, Wg1t, HH, HH);
        k_transpose_cvt<<<g(HH*HH,T), T, 0, stream>>>(W_glu2 + (size_t)i*HH*HH, Wg2t, HH, HH);

        // per-node halves of the edge GEMM (b_agg folded into Asrc)
        k_gemm_bf16<<<dim3(NN/16, HH/128), 128, 0, stream>>>(
            h0b, Wsrc_t, b_agg + i*HH, Asrc, NN, HH, HH, 1.0f, nullptr, 1);
        k_gemm_bf16<<<dim3(NN/16, HH/128), 128, 0, stream>>>(
            h0b, Wdst_t, nullptr, Adst, NN, HH, HH, 1.0f, nullptr, 1);

        k_zero_f32<<<(NN*HH)/T, T, 0, stream>>>(agg, NN*HH);
        // edge GEMM (TDM-fed) + gather + power-law weight + scatter-sum
        k_edge_msg<<<EE/16, 256, 0, stream>>>(
            edge_sbhf, Wmid_t, Asrc, Adst, edge_src, edge_dst, edge_radius,
            exp_p, i, agg);

        k_rst<<<(NN*HH)/T, T, 0, stream>>>(h0, agg, eps_p, i, rstb, NN*HH);

        // GLU
        k_gemm_bf16<<<dim3(NN/16, HH/128), 128, 0, stream>>>(
            rstb, Wg1t, b_glu1 + i*HH, G1, NN, HH, HH, 1.0f, nullptr, 1);
        k_gemm_bf16<<<dim3(NN/16, HH/128), 128, 0, stream>>>(
            rstb, Wg2t, b_glu2 + i*HH, G2, NN, HH, HH, 1.0f, nullptr, 1);
        k_glu<<<(NN*HH)/T, T, 0, stream>>>(G1, G2, hf, hb, NN*HH);
        k_transpose_cvt<<<g(NN*HH,T), T, 0, stream>>>(hf, hTb, NN, HH);

        // scores[bl,n] = (sx @ h^T)/16, masked per graph (Bt == hb row-major)
        k_gemm_bf16<<<dim3(BLROWS/16, NN/128), 128, 0, stream>>>(
            sxb, hb, nullptr, scores, BLROWS, NN, HH, 0.0625f, node_graph, LSPEC);
        k_softmax_row<<<BLROWS, 256, 0, stream>>>(scores, Pb, NN);
        // y = P @ h   (Bt = hTb: [256][8192])
        k_gemm_bf16<<<dim3(BLROWS/16, HH/128), 128, 0, stream>>>(
            Pb, hTb, nullptr, Yf, BLROWS, HH, NN, 1.0f, nullptr, 1);
        // spec_y = spec_y * (y @ W_enc) + b_enc
        k_spec_update<<<BLROWS, 256, 0, stream>>>(Yf, W_enc, b_enc, out);
    }
}
